// Swin_85014582657118
// MI455X (gfx1250) — matrix operand order repaired and checked by reading, not compiled, not run on hardware
//
#include <hip/hip_runtime.h>
#include <hip/hip_bf16.h>
#include <math.h>

// ---------------------------------------------------------------------------
// Swin-B forward for MI455X (gfx1250, wave32, WMMA).
// All matmuls lower to v_wmma_f32_16x16x32_f16 (f16 A/B, f32 accum).
// A-operand LDS staging uses gfx1250 async global->LDS copies when available.
// ---------------------------------------------------------------------------

typedef _Float16 f16;
typedef __attribute__((ext_vector_type(16))) _Float16 v16h;
typedef __attribute__((ext_vector_type(8)))  _Float16 v8h;
typedef __attribute__((ext_vector_type(8)))  float    v8f;
typedef __attribute__((ext_vector_type(4)))  float    v4f;

#define WIN 7
#define WS2 49

#if defined(__gfx1250__) && __has_builtin(__builtin_amdgcn_global_load_async_to_lds_b128) && __has_builtin(__builtin_amdgcn_s_wait_asynccnt)
#define USE_ASYNC_LDS 1
typedef __attribute__((ext_vector_type(4))) int i4;
typedef __attribute__((address_space(1))) i4* gptr_i4;   // global int4*
typedef __attribute__((address_space(3))) i4* lptr_i4;   // LDS    int4*
#else
#define USE_ASYNC_LDS 0
#endif

__device__ inline v16h cat8(v8h lo, v8h hi) {
  return __builtin_shufflevector(lo, hi, 0,1,2,3,4,5,6,7,8,9,10,11,12,13,14,15);
}
__device__ inline float wave_sum(float v) {
#pragma unroll
  for (int o = 1; o < 32; o <<= 1) v += __shfl_xor(v, o, 32);
  return v;
}

// ---------------------------------------------------------------------------
// Generic TN GEMM:  out[M,N] = A[M,K](f16) * W[N,K](f32->f16)^T  (+bias,
// +GELU, +row-aligned residual), writing f32 and/or f16 outputs.
// Block: 256 threads (8 waves), tile 128x64, K-step 64 (2 WMMA k-steps),
// wave tile 32x32 (8 WMMA per staging barrier). K must be a multiple of 64.
// ---------------------------------------------------------------------------
#define LDT 72  // padded LDS row stride (f16 elems) to spread banks

__global__ __launch_bounds__(256)
void gemm_wmma_k(const f16* __restrict__ A, const float* __restrict__ W,
                 const float* __restrict__ bias, const float* __restrict__ resid,
                 float* __restrict__ outF, f16* __restrict__ outH,
                 int M, int N, int K, int act)
{
  __shared__ f16 As[128 * LDT];
  __shared__ f16 Bs[64 * LDT];
  const int t = threadIdx.x, lane = t & 31, wid = t >> 5;
  const int wm = wid & 3, wn = wid >> 2;
  const int bm = blockIdx.x * 128, bn = blockIdx.y * 64;
  const int hh = lane >> 4, ll = lane & 15;
  const int rA = t >> 1, halfA = t & 1;   // A staging role: 32 f16 each
  const int cB = t >> 2, qB = t & 3;      // B staging role: 16 f32 each

  v8f acc[2][2] = {};

  for (int k0 = 0; k0 < K; k0 += 64) {
    __syncthreads();
    { // stage A tile 128x64 f16 (async global->LDS when available)
      int gm = bm + rA;
      f16* dst = &As[rA * LDT + halfA * 32];
      if (gm < M) {
        const f16* src = A + (size_t)gm * K + k0 + halfA * 32;
#if USE_ASYNC_LDS
#pragma unroll
        for (int u = 0; u < 4; ++u)
          __builtin_amdgcn_global_load_async_to_lds_b128(
              (gptr_i4)(src + u * 8), (lptr_i4)(dst + u * 8), 0, 0);
#else
        v8h a0 = *(const v8h*)(src);
        v8h a1 = *(const v8h*)(src + 8);
        v8h a2 = *(const v8h*)(src + 16);
        v8h a3 = *(const v8h*)(src + 24);
        *(v8h*)(dst) = a0; *(v8h*)(dst + 8) = a1;
        *(v8h*)(dst + 16) = a2; *(v8h*)(dst + 24) = a3;
#endif
      } else {
        v8h z = {};
#pragma unroll
        for (int u = 0; u < 4; ++u) *(v8h*)(dst + u * 8) = z;
      }
    }
    { // stage B tile 64x64: f32 global -> f16 LDS (16 values per thread)
      const float* src = W + (size_t)(bn + cB) * K + k0 + qB * 16;
      f16* dst = &Bs[cB * LDT + qB * 16];
      v4f a0 = *(const v4f*)src;
      v4f a1 = *(const v4f*)(src + 4);
      v4f a2 = *(const v4f*)(src + 8);
      v4f a3 = *(const v4f*)(src + 12);
#pragma unroll
      for (int i = 0; i < 4; ++i) {
        dst[i]      = (f16)a0[i];
        dst[4 + i]  = (f16)a1[i];
        dst[8 + i]  = (f16)a2[i];
        dst[12 + i] = (f16)a3[i];
      }
      if (k0 + 64 < K) __builtin_prefetch(src + 64, 0, 0);  // global_prefetch_b8
    }
#if USE_ASYNC_LDS
    __builtin_amdgcn_s_wait_asynccnt(0);
#endif
    __syncthreads();

#pragma unroll
    for (int ks = 0; ks < 2; ++ks) {
      v16h af[2], bf[2];
#pragma unroll
      for (int mt = 0; mt < 2; ++mt) {
        const f16* p = &As[(wm * 32 + mt * 16 + ll) * LDT + ks * 32];
        af[mt] = cat8(*(const v8h*)(p + hh * 8), *(const v8h*)(p + 16 + hh * 8));
      }
#pragma unroll
      for (int nt = 0; nt < 2; ++nt) {
        const f16* p = &Bs[(wn * 32 + nt * 16 + ll) * LDT + ks * 32];
        bf[nt] = cat8(*(const v8h*)(p + hh * 8), *(const v8h*)(p + 16 + hh * 8));
      }
#pragma unroll
      for (int mt = 0; mt < 2; ++mt)
#pragma unroll
        for (int nt = 0; nt < 2; ++nt)
          acc[mt][nt] = __builtin_amdgcn_wmma_f32_16x16x32_f16(
              false, af[mt], false, bf[nt], (short)0, acc[mt][nt], false, false);
    }
  }

  // epilogue
#pragma unroll
  for (int mt = 0; mt < 2; ++mt)
#pragma unroll
    for (int nt = 0; nt < 2; ++nt) {
      int n = bn + wn * 32 + nt * 16 + ll;
      float bv = bias ? bias[n] : 0.f;
#pragma unroll
      for (int i = 0; i < 8; ++i) {
        int m = bm + wm * 32 + mt * 16 + i + 8 * hh;
        if (m < M) {
          float v = acc[mt][nt][i] + bv;
          if (act) v = 0.5f * v * (1.f + erff(v * 0.70710678118f));  // exact GELU
          if (resid) v += resid[(size_t)m * N + n];
          if (outF) outF[(size_t)m * N + n] = v;
          if (outH) outH[(size_t)m * N + n] = (f16)v;
        }
      }
    }
}

// ---------------------------------------------------------------------------
// Window attention. One 128-thread block (4 waves) per (window, head).
// dh = 32 always -> QK^T is a single k-step; P@V pads K 49->64.
// rpb + shifted-window mask computed from coordinates inline.
// ---------------------------------------------------------------------------
__device__ inline int swin_region(int y, int res) {
  return y < res - WIN ? 0 : (y < res - 3 ? 1 : 2);
}

__global__ __launch_bounds__(128)
void attn_win_k(const f16* __restrict__ qkv, const float* __restrict__ rpb,
                f16* __restrict__ out, int C, int heads, int nWw, int shift, int res)
{
  const int h = blockIdx.x % heads;
  const int wIdx = blockIdx.x / heads;
  const int nW = nWw * nWw;
  const int w = wIdx % nW;
  const int wy = w / nWw, wx = w % nWw;

  __shared__ f16 Qp[64 * 32];
  __shared__ f16 Kp[64 * 32];
  __shared__ f16 Vt[32 * 64];         // V transposed: Vt[n][k]
  __shared__ f16 Pst[4 * 16 * 64];    // per-wave P tiles

  const int t = threadIdx.x, lane = t & 31, wid = t >> 5;
  const int hh = lane >> 4, ll = lane & 15;
  const size_t base = (size_t)wIdx * WS2;
  const int C3 = 3 * C;

  { // stage Q,K,V (rows padded to 64 with zeros)
    int r = t >> 1, half = t & 1;
    v8h z = {};
    v8h qa = z, qb = z, ka = z, kb = z, va = z, vb = z;
    if (r < WS2) {
      const f16* row = qkv + (base + r) * (size_t)C3;
      const f16* pq = row + h * 32 + half * 16;
      const f16* pk = row + C + h * 32 + half * 16;
      const f16* pv = row + 2 * C + h * 32 + half * 16;
      qa = *(const v8h*)pq; qb = *(const v8h*)(pq + 8);
      ka = *(const v8h*)pk; kb = *(const v8h*)(pk + 8);
      va = *(const v8h*)pv; vb = *(const v8h*)(pv + 8);
    }
    *(v8h*)&Qp[r * 32 + half * 16] = qa; *(v8h*)&Qp[r * 32 + half * 16 + 8] = qb;
    *(v8h*)&Kp[r * 32 + half * 16] = ka; *(v8h*)&Kp[r * 32 + half * 16 + 8] = kb;
#pragma unroll
    for (int i = 0; i < 8; ++i) {
      Vt[(half * 16 + i) * 64 + r] = va[i];
      Vt[(half * 16 + 8 + i) * 64 + r] = vb[i];
    }
  }
  __syncthreads();

  const int row0 = wid * 16;

  // S = Q K^T  : 4 n-tiles, single k-step (K=32)
  v16h aq;
  {
    const f16* p = &Qp[(row0 + ll) * 32];
    aq = cat8(*(const v8h*)(p + hh * 8), *(const v8h*)(p + 16 + hh * 8));
  }
  float sv[4][8];
#pragma unroll
  for (int nt = 0; nt < 4; ++nt) {
    const f16* p = &Kp[(nt * 16 + ll) * 32];
    v16h bk = cat8(*(const v8h*)(p + hh * 8), *(const v8h*)(p + 16 + hh * 8));
    v8f z = {};
    v8f s = __builtin_amdgcn_wmma_f32_16x16x32_f16(false, aq, false, bk,
                                                   (short)0, z, false, false);
    const float scale = 0.17677669529663687f;  // 1/sqrt(32)
#pragma unroll
    for (int i = 0; i < 8; ++i) {
      int m = row0 + i + 8 * hh;
      int n = nt * 16 + ll;
      float val = -1e30f;
      if (m < WS2 && n < WS2) {
        int tyi = m / WIN, txi = m % WIN, tyj = n / WIN, txj = n % WIN;
        int ridx = (tyi - tyj + WIN - 1) * (2 * WIN - 1) + (txi - txj + WIN - 1);
        val = s[i] * scale + rpb[ridx * heads + h];
        if (shift) {
          int ci = 3 * swin_region(wy * WIN + tyi, res) + swin_region(wx * WIN + txi, res);
          int cj = 3 * swin_region(wy * WIN + tyj, res) + swin_region(wx * WIN + txj, res);
          if (ci != cj) val -= 100.f;
        }
      }
      sv[nt][i] = val;
    }
  }

  // row softmax: rows live in fixed (i, lane-group) slots -> reduce over the
  // 16 lanes of each half-wave group (xor masks 1/2/4/8 stay inside a group)
  float rmax[8], rsum[8];
#pragma unroll
  for (int i = 0; i < 8; ++i) {
    float m = sv[0][i];
#pragma unroll
    for (int nt = 1; nt < 4; ++nt) m = fmaxf(m, sv[nt][i]);
    rmax[i] = m;
  }
#pragma unroll
  for (int o = 1; o < 16; o <<= 1)
#pragma unroll
    for (int i = 0; i < 8; ++i) rmax[i] = fmaxf(rmax[i], __shfl_xor(rmax[i], o, 32));
#pragma unroll
  for (int i = 0; i < 8; ++i) rsum[i] = 0.f;
#pragma unroll
  for (int nt = 0; nt < 4; ++nt)
#pragma unroll
    for (int i = 0; i < 8; ++i) {
      float e = exp2f((sv[nt][i] - rmax[i]) * 1.4426950408889634f);
      sv[nt][i] = e;
      rsum[i] += e;
    }
#pragma unroll
  for (int o = 1; o < 16; o <<= 1)
#pragma unroll
    for (int i = 0; i < 8; ++i) rsum[i] += __shfl_xor(rsum[i], o, 32);
#pragma unroll
  for (int nt = 0; nt < 4; ++nt)
#pragma unroll
    for (int i = 0; i < 8; ++i) {
      int m = i + 8 * hh;
      Pst[wid * 1024 + m * 64 + nt * 16 + ll] = (f16)(sv[nt][i] / rsum[i]);
    }
  __syncthreads();

  // O = P V : K padded to 64 (two k-steps), N = 32 (two n-tiles)
  v8f o[2] = {};
#pragma unroll
  for (int kt = 0; kt < 2; ++kt) {
    const f16* p = &Pst[wid * 1024 + ll * 64 + kt * 32];
    v16h ap = cat8(*(const v8h*)(p + hh * 8), *(const v8h*)(p + 16 + hh * 8));
#pragma unroll
    for (int nt = 0; nt < 2; ++nt) {
      const f16* q = &Vt[(nt * 16 + ll) * 64 + kt * 32];
      v16h bv = cat8(*(const v8h*)(q + hh * 8), *(const v8h*)(q + 16 + hh * 8));
      o[nt] = __builtin_amdgcn_wmma_f32_16x16x32_f16(false, ap, false, bv,
                                                     (short)0, o[nt], false, false);
    }
  }
#pragma unroll
  for (int nt = 0; nt < 2; ++nt)
#pragma unroll
    for (int i = 0; i < 8; ++i) {
      int m = row0 + i + 8 * hh;
      int n = nt * 16 + ll;
      if (m < WS2) out[(base + m) * (size_t)C + h * 32 + n] = (f16)o[nt][i];
    }
}

// ---------------------------------------------------------------------------
// Patch embed: 4x4/s4 conv (K=48) + bias + LN over C=128. One block per token.
// ---------------------------------------------------------------------------
__global__ __launch_bounds__(128)
void patch_embed_k(const float* __restrict__ img, const float* __restrict__ w,
                   const float* __restrict__ bias, const float* __restrict__ ng,
                   const float* __restrict__ nb, float* __restrict__ x)
{
  const int row = blockIdx.x;       // b*3136 + py*56 + px
  const int c = threadIdx.x;        // channel 0..127
  const int bb = row / 3136, p = row % 3136, py = p / 56, px = p % 56;
  __shared__ float patch[48];
  __shared__ float red[128];
  if (c < 48) {
    int ci = c / 16, k = c % 16, ky = k / 4, kx = k % 4;
    patch[c] = img[((size_t)(bb * 3 + ci) * 224 + py * 4 + ky) * 224 + px * 4 + kx];
  }
  __syncthreads();
  float s = bias[c];
#pragma unroll
  for (int i = 0; i < 48; ++i) s += patch[i] * w[c * 48 + i];
  red[c] = s; __syncthreads();
  for (int o = 64; o > 0; o >>= 1) { if (c < o) red[c] += red[c + o]; __syncthreads(); }
  float mu = red[0] / 128.f; __syncthreads();
  float d = s - mu;
  red[c] = d * d; __syncthreads();
  for (int o = 64; o > 0; o >>= 1) { if (c < o) red[c] += red[c + o]; __syncthreads(); }
  float rstd = rsqrtf(red[0] / 128.f + 1e-5f);
  x[(size_t)row * 128 + c] = d * rstd * ng[c] + nb[c];
}

// ---------------------------------------------------------------------------
// LayerNorm variants (one wave per output row), emitting f16 GEMM inputs.
// ---------------------------------------------------------------------------
__global__ __launch_bounds__(256)
void ln_part_k(const float* __restrict__ x, const float* __restrict__ g,
               const float* __restrict__ b, f16* __restrict__ out,
               int B_, int res, int C, int shift)   // LN + roll + window part.
{
  const int lane = threadIdx.x & 31;
  const int wrow = blockIdx.x * 8 + (threadIdx.x >> 5);
  const int rows = B_ * res * res;
  if (wrow >= rows) return;
  const int nWw = res / WIN, nW = nWw * nWw;
  const int tok = wrow % WS2, wIdx = wrow / WS2;
  const int w = wIdx % nW, bb = wIdx / nW;
  const int wy = w / nWw, wx = w % nWw, ty = tok / WIN, tx = tok % WIN;
  const int y = (wy * WIN + ty + shift) % res;
  const int xx = (wx * WIN + tx + shift) % res;
  const float* src = x + ((size_t)bb * res * res + y * res + xx) * C;
  float s = 0.f, s2 = 0.f;
  for (int c = lane; c < C; c += 32) { float v = src[c]; s += v; s2 += v * v; }
  s = wave_sum(s); s2 = wave_sum(s2);
  float mu = s / C, var = s2 / C - mu * mu, rstd = rsqrtf(var + 1e-5f);
  f16* dst = out + (size_t)wrow * C;
  for (int c = lane; c < C; c += 32) dst[c] = (f16)((src[c] - mu) * rstd * g[c] + b[c]);
}

__global__ __launch_bounds__(256)
void ln_plain_k(const float* __restrict__ x, const float* __restrict__ g,
                const float* __restrict__ b, f16* __restrict__ out, int rows, int C)
{
  const int lane = threadIdx.x & 31;
  const int r = blockIdx.x * 8 + (threadIdx.x >> 5);
  if (r >= rows) return;
  const float* src = x + (size_t)r * C;
  float s = 0.f, s2 = 0.f;
  for (int c = lane; c < C; c += 32) { float v = src[c]; s += v; s2 += v * v; }
  s = wave_sum(s); s2 = wave_sum(s2);
  float mu = s / C, rstd = rsqrtf(s2 / C - mu * mu + 1e-5f);
  f16* dst = out + (size_t)r * C;
  for (int c = lane; c < C; c += 32) dst[c] = (f16)((src[c] - mu) * rstd * g[c] + b[c]);
}

__global__ __launch_bounds__(256)
void ln_merge_k(const float* __restrict__ x, const float* __restrict__ g,
                const float* __restrict__ b, f16* __restrict__ out,
                int B_, int res, int C)   // 2x2 gather-concat + LN over 4C
{
  const int lane = threadIdx.x & 31;
  const int half = res / 2;
  const int r = blockIdx.x * 8 + (threadIdx.x >> 5);
  if (r >= B_ * half * half) return;
  const int ox = r % half, oy = (r / half) % half, bb = r / (half * half);
  const int C4 = 4 * C;
  float s = 0.f, s2 = 0.f;
  for (int c4 = lane; c4 < C4; c4 += 32) {
    int part = c4 / C, c = c4 % C, dy = part & 1, dx = part >> 1;
    float v = x[((size_t)(bb * res + 2 * oy + dy) * res + 2 * ox + dx) * C + c];
    s += v; s2 += v * v;
  }
  s = wave_sum(s); s2 = wave_sum(s2);
  float mu = s / C4, rstd = rsqrtf(s2 / C4 - mu * mu + 1e-5f);
  f16* dst = out + (size_t)r * C4;
  for (int c4 = lane; c4 < C4; c4 += 32) {
    int part = c4 / C, c = c4 % C, dy = part & 1, dx = part >> 1;
    float v = x[((size_t)(bb * res + 2 * oy + dy) * res + 2 * ox + dx) * C + c];
    dst[c4] = (f16)((v - mu) * rstd * g[c4] + b[c4]);
  }
}

// window reverse + un-roll + residual add (x += reverse(tmp))
__global__ __launch_bounds__(256)
void rev_add_k(float* __restrict__ x, const float* __restrict__ tmp,
               int B_, int res, int C, int shift)
{
  size_t idx = (size_t)blockIdx.x * 256 + threadIdx.x;
  size_t total = (size_t)B_ * res * res * C;
  if (idx >= total) return;
  int c = idx % C;
  size_t r = idx / C;
  int xc = r % res, y = (r / res) % res, bb = r / ((size_t)res * res);
  int yr = (y - shift + res) % res;
  int xr = (xc - shift + res) % res;
  int nWw = res / WIN;
  int wrow = (((bb * nWw + yr / WIN) * nWw + xr / WIN) * WS2) + (yr % WIN) * WIN + (xr % WIN);
  x[idx] += tmp[(size_t)wrow * C + c];
}

// final LN stats (one wave per row) + mean-pool with LN applied
__global__ __launch_bounds__(256)
void ln_stats_k(const float* __restrict__ x, float* __restrict__ st, int rows, int C)
{
  const int lane = threadIdx.x & 31;
  const int r = blockIdx.x * 8 + (threadIdx.x >> 5);
  if (r >= rows) return;
  const float* src = x + (size_t)r * C;
  float s = 0.f, s2 = 0.f;
  for (int c = lane; c < C; c += 32) { float v = src[c]; s += v; s2 += v * v; }
  s = wave_sum(s); s2 = wave_sum(s2);
  if (lane == 0) {
    float mu = s / C;
    st[2 * r] = mu;
    st[2 * r + 1] = rsqrtf(s2 / C - mu * mu + 1e-5f);
  }
}

__global__ __launch_bounds__(256)
void pool_out_k(const float* __restrict__ x, const float* __restrict__ st,
                const float* __restrict__ g, const float* __restrict__ b,
                float* __restrict__ out)
{
  const int bb = blockIdx.x;
  const int c = blockIdx.y * 256 + threadIdx.x;   // C = 1024
  float acc = 0.f;
  for (int t = 0; t < WS2; ++t) {
    int r = bb * WS2 + t;
    acc += (x[(size_t)r * 1024 + c] - st[2 * r]) * st[2 * r + 1];
  }
  out[bb * 1024 + c] = acc * (1.f / WS2) * g[c] + b[c];
}

// ---------------------------------------------------------------------------
// Host driver
// ---------------------------------------------------------------------------
struct BlkP {
  const float *n1g, *n1b, *qkvw, *qkvb, *projw, *projb, *rpb,
              *n2g, *n2b, *fc1w, *fc1b, *fc2w, *fc2b;
};
struct MrgP { const float *ng, *nb, *rw; };

extern "C" void kernel_launch(void* const* d_in, const int* in_sizes, int n_in,
                              void* d_out, int out_size, void* d_ws, size_t ws_size,
                              hipStream_t stream)
{
  (void)in_sizes; (void)n_in; (void)out_size; (void)ws_size;
  static const int DEPTHS[4] = {2, 2, 18, 2};
  static const int HEADS[4]  = {4, 8, 16, 32};
  const int B = 8;

  // inputs in setup_inputs() dict (insertion) order
  int ix = 0;
  const float* img   = (const float*)d_in[ix++];
  const float* pe_w  = (const float*)d_in[ix++];
  const float* pe_b  = (const float*)d_in[ix++];
  const float* pe_ng = (const float*)d_in[ix++];
  const float* pe_nb = (const float*)d_in[ix++];
  BlkP blks[4][18];
  MrgP mrg[3];
  for (int s = 0; s < 4; ++s) {
    for (int i = 0; i < DEPTHS[s]; ++i) {
      BlkP& p = blks[s][i];
      p.n1g = (const float*)d_in[ix++];  p.n1b  = (const float*)d_in[ix++];
      p.qkvw = (const float*)d_in[ix++]; p.qkvb = (const float*)d_in[ix++];
      p.projw = (const float*)d_in[ix++];p.projb= (const float*)d_in[ix++];
      p.rpb = (const float*)d_in[ix++];
      p.n2g = (const float*)d_in[ix++];  p.n2b  = (const float*)d_in[ix++];
      p.fc1w = (const float*)d_in[ix++]; p.fc1b = (const float*)d_in[ix++];
      p.fc2w = (const float*)d_in[ix++]; p.fc2b = (const float*)d_in[ix++];
    }
    if (s < 3) {
      mrg[s].ng = (const float*)d_in[ix++];
      mrg[s].nb = (const float*)d_in[ix++];
      mrg[s].rw = (const float*)d_in[ix++];
    }
  }
  const float* norm_g = (const float*)d_in[ix++];
  const float* norm_b = (const float*)d_in[ix++];

  // workspace carve-out (~58 MB)
  auto au = [](size_t v) { return (v + 255) & ~(size_t)255; };
  char* p = (char*)d_ws;
  const size_t MAXF32 = 25088ull * 128;        // max B*L*C
  const size_t MAXF16 = 25088ull * 512;        // max B*L*4C
  float* xbuf  = (float*)p; p += au(MAXF32 * 4);
  float* tbuf  = (float*)p; p += au(MAXF32 * 4);
  f16*   h16a  = (f16*)p;   p += au(MAXF32 * 2);
  f16*   h16b  = (f16*)p;   p += au(MAXF16 * 2);
  float* stats = (float*)p; p += au(392ull * 2 * 4);

  // patch embed + LN -> x f32 [B, 3136, 128]
  patch_embed_k<<<B * 3136, 128, 0, stream>>>(img, pe_w, pe_b, pe_ng, pe_nb, xbuf);

  int C = 128, res = 56;
  for (int s = 0; s < 4; ++s) {
    const int L = res * res, M = B * L, hds = HEADS[s];
    const int nWw = res / WIN, nwin = B * nWw * nWw;
    for (int i = 0; i < DEPTHS[s]; ++i) {
      const BlkP& bp = blks[s][i];
      const int shift = (i % 2 == 1 && res > WIN) ? WIN / 2 : 0;

      // LN1 + roll + window partition -> f16
      ln_part_k<<<M / 8, 256, 0, stream>>>(xbuf, bp.n1g, bp.n1b, h16a, B, res, C, shift);
      // qkv = xw @ qkvw^T + qkvb (f16 out)
      gemm_wmma_k<<<dim3((M + 127) / 128, 3 * C / 64), 256, 0, stream>>>(
          h16a, bp.qkvw, bp.qkvb, nullptr, nullptr, h16b, M, 3 * C, C, 0);
      // window attention (WMMA QK^T + softmax + WMMA PV)
      attn_win_k<<<nwin * hds, 128, 0, stream>>>(h16b, bp.rpb, h16a, C, hds, nWw, shift, res);
      // proj
      gemm_wmma_k<<<dim3((M + 127) / 128, C / 64), 256, 0, stream>>>(
          h16a, bp.projw, bp.projb, nullptr, tbuf, nullptr, M, C, C, 0);
      // window reverse + un-roll + residual
      rev_add_k<<<(int)(((size_t)M * C + 255) / 256), 256, 0, stream>>>(
          xbuf, tbuf, B, res, C, shift);
      // LN2 -> f16
      ln_plain_k<<<M / 8, 256, 0, stream>>>(xbuf, bp.n2g, bp.n2b, h16a, M, C);
      // fc1 + exact GELU (f16 out)
      gemm_wmma_k<<<dim3((M + 127) / 128, 4 * C / 64), 256, 0, stream>>>(
          h16a, bp.fc1w, bp.fc1b, nullptr, nullptr, h16b, M, 4 * C, C, 1);
      // fc2 + residual -> x (f32)
      gemm_wmma_k<<<dim3((M + 127) / 128, C / 64), 256, 0, stream>>>(
          h16b, bp.fc2w, bp.fc2b, xbuf, xbuf, nullptr, M, C, 4 * C, 0);
    }
    if (s < 3) {
      const int half = res / 2, Mm = B * half * half;
      ln_merge_k<<<(Mm + 7) / 8, 256, 0, stream>>>(xbuf, mrg[s].ng, mrg[s].nb,
                                                   h16b, B, res, C);
      gemm_wmma_k<<<dim3((Mm + 127) / 128, 2 * C / 64), 256, 0, stream>>>(
          h16b, mrg[s].rw, nullptr, nullptr, xbuf, nullptr, Mm, 2 * C, 4 * C, 0);
      C *= 2; res /= 2;
    }
  }

  // final LN + global average pool -> [8, 1024]
  ln_stats_k<<<392 / 8, 256, 0, stream>>>(xbuf, stats, 392, 1024);
  pool_out_k<<<dim3(8, 4), 256, 0, stream>>>(xbuf, stats, norm_g, norm_b, (float*)d_out);
}
